// DistanceGatFC_41884521071136
// MI455X (gfx1250) — compile-verified, matrix operand order checked by reading, not run on hardware
//
#include <hip/hip_runtime.h>
#include <hip/hip_bf16.h>

// ---- problem constants (shapes are fixed by the reference) ----
#define RAW       12
#define GPS       32
#define FEAT      74      // 2*GPS + (RAW-2)
#define KPAD      96      // FEAT padded to 3 WMMA K-steps of 32
#define H_HEADS   4
#define F_DIM     32
#define HFDIM     128     // H*F
#define NEG_SLOPE 0.2f

typedef __attribute__((ext_vector_type(16))) _Float16 v16h;
typedef __attribute__((ext_vector_type(8)))  float    v8f;

union HFrag { uint4 q[2]; v16h v; };

// monotonic encoding of float into uint so atomicMax(uint) == float max
__device__ __forceinline__ unsigned fenc(float f) {
    unsigned u = __float_as_uint(f);
    return (u & 0x80000000u) ? ~u : (u | 0x80000000u);
}
__device__ __forceinline__ float fdec(unsigned u) {
    unsigned b = (u & 0x80000000u) ? (u ^ 0x80000000u) : ~u;
    return __uint_as_float(b);
}

// ---------------------------------------------------------------------------
// 1) Build f16 GAT input features [Npad, 96]: [lon_emb | lat_emb | raw[:10] | 0-pad]
// ---------------------------------------------------------------------------
__global__ void k_build_feat(const float* __restrict__ nf,
                             const float* __restrict__ lonT,
                             const float* __restrict__ latT,
                             _Float16* __restrict__ featH, int N)
{
    const int n = blockIdx.x;
    const int c = threadIdx.x;
    if (c >= KPAD) return;
    float v = 0.f;
    if (n < N) {
        if (c < GPS) {
            int li = (int)nf[(size_t)n * RAW + (RAW - 2)];
            v = lonT[(size_t)li * GPS + c];
        } else if (c < 2 * GPS) {
            int li = (int)nf[(size_t)n * RAW + (RAW - 1)];
            v = latT[(size_t)li * GPS + (c - GPS)];
        } else if (c < FEAT) {
            v = nf[(size_t)n * RAW + (c - 2 * GPS)];
        }
    }
    featH[(size_t)n * KPAD + c] = (_Float16)v;
}

// ---------------------------------------------------------------------------
// 2) Pre-swizzle [W_proj | W_skip] (f32 [74,128] each) into f16 WMMA
//    B-fragments: frag(t, s, lane) = 16 contiguous halves.
//    K mapping per lane (hi = lane>>4): h<8 -> K = hi*8+h ; h>=8 -> K = 16+hi*8+(h-8)
// ---------------------------------------------------------------------------
__global__ void k_build_w(const float* __restrict__ Wp,
                          const float* __restrict__ Ws,
                          _Float16* __restrict__ Wcp)
{
    const int idx = blockIdx.x * blockDim.x + threadIdx.x;   // (t*3+s)*32 + lane
    if (idx >= 16 * 3 * 32) return;
    const int lane = idx & 31;
    const int s    = (idx >> 5) % 3;
    const int t    = idx / 96;
    const int hi   = lane >> 4;
    const int col  = t * 16 + (lane & 15);                   // 0..255
    _Float16* dst = Wcp + (size_t)idx * 16;
    #pragma unroll
    for (int h = 0; h < 16; ++h) {
        int kl = (h < 8) ? (hi * 8 + h) : (16 + hi * 8 + (h - 8));
        int K  = s * 32 + kl;
        float v = 0.f;
        if (K < FEAT)
            v = (col < HFDIM) ? Wp[(size_t)K * HFDIM + col]
                              : Ws[(size_t)K * HFDIM + (col - HFDIM)];
        dst[h] = (_Float16)v;
    }
}

// ---------------------------------------------------------------------------
// 3) Fused GEMM: [16 x 96] tile of features times [96 x 256] weights via WMMA.
//    Output cols 0..127 -> proj, cols 128..255 -> outacc (skip connection,
//    which seeds the message accumulator).
//    Block = 256 threads = 8 waves; wave w computes col-tiles 2w, 2w+1.
// ---------------------------------------------------------------------------
__global__ __launch_bounds__(256) void k_gemm(const _Float16* __restrict__ featH,
                                              const _Float16* __restrict__ Wcp,
                                              float* __restrict__ proj,
                                              float* __restrict__ outacc)
{
    __shared__ __align__(16) _Float16 Atile[16 * KPAD];      // 3 KB
    const int m0 = blockIdx.x * 16;

    // tile rows are contiguous in memory: straight 768-dword copy to LDS
    const uint* gsrc = (const uint*)(featH + (size_t)m0 * KPAD);
    uint* lds_u = (uint*)Atile;
    for (int i = threadIdx.x; i < (16 * KPAD) / 2; i += 256) lds_u[i] = gsrc[i];
    __syncthreads();

    const int lane = threadIdx.x & 31;
    const int wave = threadIdx.x >> 5;
    const int hi   = lane >> 4;
    const int r    = lane & 15;

    // A fragments for the 3 K-steps (ds_load_b128 x2 each)
    HFrag a[3];
    #pragma unroll
    for (int s = 0; s < 3; ++s) {
        const _Float16* rowp = Atile + r * KPAD + s * 32 + hi * 8;
        a[s].q[0] = *(const uint4*)(rowp);        // K = s*32 + hi*8 + 0..7
        a[s].q[1] = *(const uint4*)(rowp + 16);   // K = s*32 + 16 + hi*8 + 0..7
    }

    #pragma unroll
    for (int tt = 0; tt < 2; ++tt) {
        const int t = wave * 2 + tt;
        v8f acc = {0.f, 0.f, 0.f, 0.f, 0.f, 0.f, 0.f, 0.f};
        #pragma unroll
        for (int s = 0; s < 3; ++s) {
            HFrag b;
            const _Float16* bp = Wcp + ((size_t)(t * 3 + s) * 32 + lane) * 16;
            b.q[0] = *(const uint4*)(bp);
            b.q[1] = *(const uint4*)(bp + 8);
            acc = __builtin_amdgcn_wmma_f32_16x16x32_f16(
                false, a[s].v, false, b.v, (short)0, acc, false, false);
        }
        // C/D layout: lane -> col (lane&15), vgpr v -> row v + hi*8
        const int n = lane & 15;
        float* dst;
        int col;
        if (t < 8) { dst = proj;   col = t * 16 + n; }
        else       { dst = outacc; col = (t - 8) * 16 + n; }
        #pragma unroll
        for (int v = 0; v < 8; ++v) {
            int row = m0 + v + hi * 8;
            dst[(size_t)row * HFDIM + col] = acc[v];
        }
    }
}

// ---------------------------------------------------------------------------
// 4) Attention scores s_src/s_tgt = <proj[n,h,:], a_src/a_tgt[h,:]>
// ---------------------------------------------------------------------------
__global__ void k_scores(const float* __restrict__ proj,
                         const float* __restrict__ a_src,
                         const float* __restrict__ a_tgt,
                         float* __restrict__ ssrc, float* __restrict__ stgt, int N)
{
    const int idx = blockIdx.x * blockDim.x + threadIdx.x;
    if (idx >= N * H_HEADS) return;
    const int n = idx >> 2, h = idx & 3;
    const float* p  = proj  + (size_t)n * HFDIM + h * F_DIM;
    const float* as = a_src + h * F_DIM;
    const float* at = a_tgt + h * F_DIM;
    float s0 = 0.f, s1 = 0.f;
    #pragma unroll
    for (int f = 0; f < F_DIM; ++f) { float pv = p[f]; s0 += pv * as[f]; s1 += pv * at[f]; }
    ssrc[idx] = s0;
    stgt[idx] = s1;
}

__device__ __forceinline__ float edge_logit(const float* ssrc, const float* stgt,
                                            int s, int t, int h)
{
    float x = ssrc[s * 4 + h] + stgt[t * 4 + h];
    return (x >= 0.f) ? x : NEG_SLOPE * x;
}

// 5) segment max over incoming edges (monotonic-uint atomicMax)
__global__ void k_edge_max(const int* __restrict__ src, const int* __restrict__ tgt,
                           const float* __restrict__ ssrc, const float* __restrict__ stgt,
                           unsigned* __restrict__ menc, int E)
{
    const int e = blockIdx.x * blockDim.x + threadIdx.x;
    if (e >= E) return;
    const int s = src[e], t = tgt[e];
    #pragma unroll
    for (int h = 0; h < H_HEADS; ++h)
        atomicMax(&menc[t * 4 + h], fenc(edge_logit(ssrc, stgt, s, t, h)));
}

// 6) segment sum of exp(e - max)
__global__ void k_edge_denom(const int* __restrict__ src, const int* __restrict__ tgt,
                             const float* __restrict__ ssrc, const float* __restrict__ stgt,
                             const unsigned* __restrict__ menc, float* __restrict__ denom, int E)
{
    const int e = blockIdx.x * blockDim.x + threadIdx.x;
    if (e >= E) return;
    const int s = src[e], t = tgt[e];
    #pragma unroll
    for (int h = 0; h < H_HEADS; ++h) {
        float x = edge_logit(ssrc, stgt, s, t, h);
        atomicAdd(&denom[t * 4 + h], __expf(x - fdec(menc[t * 4 + h])));
    }
}

// 7) weighted message scatter: one wave per edge, lane = feature
__global__ __launch_bounds__(256) void k_edge_msg(const int* __restrict__ src,
                                                  const int* __restrict__ tgt,
                                                  const float* __restrict__ ssrc,
                                                  const float* __restrict__ stgt,
                                                  const unsigned* __restrict__ menc,
                                                  const float* __restrict__ denom,
                                                  const float* __restrict__ proj,
                                                  float* __restrict__ outacc, int E)
{
    const int e = blockIdx.x * (blockDim.x >> 5) + (threadIdx.x >> 5);
    const int lane = threadIdx.x & 31;
    if (e >= E) return;
    const int s = src[e], t = tgt[e];
    #pragma unroll
    for (int h = 0; h < H_HEADS; ++h) {
        float x = edge_logit(ssrc, stgt, s, t, h);
        float a = __expf(x - fdec(menc[t * 4 + h])) / (denom[t * 4 + h] + 1e-16f);
        float val = proj[(size_t)s * HFDIM + h * F_DIM + lane] * a;
        atomicAdd(&outacc[(size_t)t * HFDIM + h * F_DIM + lane], val);
    }
}

// 8) head mean + bias + ELU -> enc[N,32]
__global__ void k_finalize(const float* __restrict__ outacc,
                           const float* __restrict__ bias,
                           float* __restrict__ enc, int N)
{
    const int idx = blockIdx.x * blockDim.x + threadIdx.x;
    if (idx >= N * F_DIM) return;
    const int n = idx >> 5, f = idx & 31;
    const float* o = outacc + (size_t)n * HFDIM;
    float x = (o[f] + o[F_DIM + f] + o[2 * F_DIM + f] + o[3 * F_DIM + f]) * 0.25f + bias[f];
    enc[idx] = (x > 0.f) ? x : (__expf(x) - 1.f);
}

// 9) pair MLP head
__global__ void k_pairs(const int* __restrict__ fromN, const int* __restrict__ toN,
                        const float* __restrict__ dist, const float* __restrict__ enc,
                        const float* __restrict__ eW, const float* __restrict__ eB,
                        const float* __restrict__ dW, const float* __restrict__ dB,
                        const float* __restrict__ oW, const float* __restrict__ oB,
                        float* __restrict__ out, int B)
{
    const int b = blockIdx.x * blockDim.x + threadIdx.x;
    if (b >= B) return;
    const float* xf = enc + (size_t)fromN[b] * F_DIM;
    const float* xt = enc + (size_t)toN[b] * F_DIM;
    const float d0 = dist[b];
    float acc = oB[0];
    for (int j = 0; j < 10; ++j) {
        float s = eB[j];
        #pragma unroll
        for (int i = 0; i < F_DIM; ++i) s += xf[i] * eW[i * 10 + j];
        #pragma unroll
        for (int i = 0; i < F_DIM; ++i) s += xt[i] * eW[(F_DIM + i) * 10 + j];
        s = fmaxf(s, 0.f);
        float d = fmaxf(d0 * dW[j] + dB[j], 0.f);
        acc += s * oW[j] + d * oW[10 + j];
    }
    out[b] = acc;
}

// ---------------------------------------------------------------------------
extern "C" void kernel_launch(void* const* d_in, const int* in_sizes, int n_in,
                              void* d_out, int out_size, void* d_ws, size_t ws_size,
                              hipStream_t stream)
{
    const int*   from_node = (const int*)  d_in[0];
    const int*   to_node   = (const int*)  d_in[1];
    const float* distance  = (const float*)d_in[2];
    const float* node_feat = (const float*)d_in[3];
    const int*   edge_src  = (const int*)  d_in[4];
    const int*   edge_tgt  = (const int*)  d_in[5];
    const float* lon_table = (const float*)d_in[6];
    const float* lat_table = (const float*)d_in[7];
    const float* W_proj    = (const float*)d_in[8];
    const float* a_src     = (const float*)d_in[9];
    const float* a_tgt     = (const float*)d_in[10];
    const float* W_skip    = (const float*)d_in[11];
    const float* gat_bias  = (const float*)d_in[12];
    const float* emb_fc_W  = (const float*)d_in[13];
    const float* emb_fc_b  = (const float*)d_in[14];
    const float* dis_fc_W  = (const float*)d_in[15];
    const float* dis_fc_b  = (const float*)d_in[16];
    const float* out_fc_W  = (const float*)d_in[17];
    const float* out_fc_b  = (const float*)d_in[18];

    const int N    = in_sizes[3] / RAW;     // 100000
    const int E    = in_sizes[4];           // 1600000
    const int B    = in_sizes[0];           // 16384
    const int Npad = (N + 15) & ~15;

    char* ws = (char*)d_ws;
    size_t off = 0;
    auto alloc = [&](size_t bytes) -> char* {
        char* p = ws + off;
        off += (bytes + 255) & ~(size_t)255;
        return p;
    };
    _Float16* featH  = (_Float16*)alloc((size_t)Npad * KPAD * sizeof(_Float16));   // 19.2 MB
    _Float16* Wcp    = (_Float16*)alloc((size_t)16 * 3 * 32 * 16 * sizeof(_Float16)); // 48 KB
    float*    proj   = (float*)   alloc((size_t)Npad * HFDIM * sizeof(float));     // 51.2 MB
    float*    outacc = (float*)   alloc((size_t)Npad * HFDIM * sizeof(float));     // 51.2 MB
    float*    ssrc   = (float*)   alloc((size_t)N * 4 * sizeof(float));
    float*    stgt   = (float*)   alloc((size_t)N * 4 * sizeof(float));
    unsigned* menc   = (unsigned*)alloc((size_t)N * 4 * sizeof(unsigned));
    float*    denom  = (float*)   alloc((size_t)N * 4 * sizeof(float));
    float*    encb   = (float*)   alloc((size_t)N * F_DIM * sizeof(float));        // 12.8 MB
    (void)ws_size; (void)n_in; (void)out_size;

    hipMemsetAsync(menc,  0, (size_t)N * 4 * sizeof(unsigned), stream);
    hipMemsetAsync(denom, 0, (size_t)N * 4 * sizeof(float),    stream);

    k_build_feat<<<Npad, 96, 0, stream>>>(node_feat, lon_table, lat_table, featH, N);
    k_build_w<<<(16 * 3 * 32 + 255) / 256, 256, 0, stream>>>(W_proj, W_skip, Wcp);
    k_gemm<<<Npad / 16, 256, 0, stream>>>(featH, Wcp, proj, outacc);
    k_scores<<<(N * H_HEADS + 255) / 256, 256, 0, stream>>>(proj, a_src, a_tgt, ssrc, stgt, N);
    k_edge_max<<<(E + 255) / 256, 256, 0, stream>>>(edge_src, edge_tgt, ssrc, stgt, menc, E);
    k_edge_denom<<<(E + 255) / 256, 256, 0, stream>>>(edge_src, edge_tgt, ssrc, stgt, menc, denom, E);
    k_edge_msg<<<(E + 7) / 8, 256, 0, stream>>>(edge_src, edge_tgt, ssrc, stgt, menc, denom,
                                                proj, outacc, E);
    k_finalize<<<(N * F_DIM + 255) / 256, 256, 0, stream>>>(outacc, gat_bias, encb, N);
    k_pairs<<<(B + 255) / 256, 256, 0, stream>>>(from_node, to_node, distance, encb,
                                                 emb_fc_W, emb_fc_b, dis_fc_W, dis_fc_b,
                                                 out_fc_W, out_fc_b, (float*)d_out, B);
}
